// SequenceTaggle_32624571580908
// MI455X (gfx1250) — matrix operand order repaired and checked by reading, not compiled, not run on hardware
//
#include <hip/hip_runtime.h>
#include <hip/hip_bf16.h>

typedef __attribute__((ext_vector_type(16))) _Float16 v16h;
typedef __attribute__((ext_vector_type(8)))  _Float16 v8h;
typedef __attribute__((ext_vector_type(8)))  float    v8f;

#define S_  96
#define T_  32
#define B_  32
#define N_  (S_*B_)    /* 3072 */
#define H_  256

__device__ __forceinline__ float sigmoidf_(float x) { return 1.0f / (1.0f + __expf(-x)); }

// ---- WMMA fragment loaders (CDNA5 16x16x32 f16 layouts, ISA 7.12.2) ----
// A matrix 16x32 (MxK), row-major source with leading dim ld (halfs).
// lane<16: M=lane, K={0..7,16..23}; lane>=16: M=lane-16, K={8..15,24..31}.
__device__ __forceinline__ v16h load_fragA(const _Float16* p, int ld, int lane) {
  int r  = lane & 15;
  int kb = (lane < 16) ? 0 : 8;
  const _Float16* q = p + (size_t)r * ld + kb;
  v8h lo = *(const v8h*)(q);        // K = kb .. kb+7
  v8h hi = *(const v8h*)(q + 16);   // K = kb+16 .. kb+23
  v16h o;
#pragma unroll
  for (int i = 0; i < 8; ++i) { o[i] = lo[i]; o[8 + i] = hi[i]; }
  return o;
}

// B matrix 32x16 (KxN). We compute C = X * W^T, so B column n == row n of the
// row-major weight W (ld halfs). lane<16: N=lane, K={0..7,16..23};
// lane>=16: N=lane-16, K={8..15,24..31}.
__device__ __forceinline__ v16h load_fragB(const _Float16* w, int ld, int lane) {
  int n  = lane & 15;
  int kb = (lane < 16) ? 0 : 8;
  const _Float16* q = w + (size_t)n * ld + kb;
  v8h lo = *(const v8h*)(q);
  v8h hi = *(const v8h*)(q + 16);
  v16h o;
#pragma unroll
  for (int i = 0; i < 8; ++i) { o[i] = lo[i]; o[8 + i] = hi[i]; }
  return o;
}

// ---- generic f32 -> f16 cast ----
__global__ void cast_kernel(const float* __restrict__ in, _Float16* __restrict__ out, int n) {
  int i = blockIdx.x * 256 + threadIdx.x;
  if (i < n) out[i] = (_Float16)in[i];
}

// ---- embedding gather: x[t, s*B+b, e] = emb[tokens[s,t,b], e]  (f16 out) ----
__global__ void embed_kernel(const int* __restrict__ tokens, const float* __restrict__ emb,
                             _Float16* __restrict__ x) {
  int id = blockIdx.x * 256 + threadIdx.x;     // over T*N*256
  int e = id & 255;
  int n = (id >> 8) % N_;
  int t = id / (N_ * 256);
  int s = n / B_, b = n % B_;
  int tok = tokens[(s * T_ + t) * B_ + b];
  x[id] = (_Float16)emb[(size_t)tok * 256 + e];
}

// ---- fused GRU scan: per block = 16 batch rows, loops over T steps.
// Computes [x*Wi^T + bi | h*Wh^T + bh] with WMMA, keeps h in LDS (f16),
// writes y (f16) at column offset col_off of a ycols-wide buffer.
// K is always 256. Gate cols: r=[0,256) z=[256,512) n=[512,768).
// LDS g: 16 x 1024 f32 -> cols 0-255 (xr+hr+b), 256-511 (xz+hz+b),
//        512-767 (hn+bh), 768-1023 (xn+bi).
__global__ void gru_scan(const _Float16* __restrict__ x,
                         const _Float16* __restrict__ Wi,
                         const _Float16* __restrict__ Wh,
                         const float* __restrict__ bi,
                         const float* __restrict__ bh,
                         _Float16* __restrict__ y,
                         int T, int N, int ycols, int col_off, int rev) {
  extern __shared__ __align__(16) char smem[];
  float*    gS = (float*)smem;                      // 16*1024 f32 = 64 KB
  _Float16* hx = (_Float16*)(smem + 16 * 1024 * 4); // 16*256  f16 =  8 KB

  const int lane = threadIdx.x & 31;
  const int wave = threadIdx.x >> 5;
  const int n0   = blockIdx.x * 16;
  const int c    = threadIdx.x;                     // 0..255

#pragma unroll
  for (int m = 0; m < 16; ++m) hx[m * 256 + c] = (_Float16)0.0f;
  __syncthreads();

  const int nlane = lane & 15;
  const int mb    = (lane < 16) ? 0 : 8;

  for (int step = 0; step < T; ++step) {
    int tt = rev ? (T - 1 - step) : step;
    const _Float16* xrow = x + ((size_t)tt * N + n0) * 256;

    // projection phase: 48 n-tiles over 8 waves (6 each)
    for (int j = 0; j < 6; ++j) {
      int nt = wave * 6 + j;
      const _Float16* wi = Wi + (size_t)nt * 16 * 256;
      const _Float16* wh = Wh + (size_t)nt * 16 * 256;
      if (nt < 32) {                 // r / z gates: fuse x and h projections
        v8f acc = {};
#pragma unroll
        for (int k = 0; k < 256; k += 32) {
          v16h a = load_fragA(xrow + k, 256, lane);
          v16h b = load_fragB(wi + k, 256, lane);
          acc = __builtin_amdgcn_wmma_f32_16x16x32_f16(false, a, false, b, (short)0, acc, false, false);
        }
#pragma unroll
        for (int k = 0; k < 256; k += 32) {
          v16h a = load_fragA(hx + k, 256, lane);
          v16h b = load_fragB(wh + k, 256, lane);
          acc = __builtin_amdgcn_wmma_f32_16x16x32_f16(false, a, false, b, (short)0, acc, false, false);
        }
        float bsum = bi[nt * 16 + nlane] + bh[nt * 16 + nlane];
#pragma unroll
        for (int i = 0; i < 8; ++i)
          gS[(mb + i) * 1024 + nt * 16 + nlane] = acc[i] + bsum;
      } else {                       // n gate: keep xn and hn separate
        v8f aX = {}, aH = {};
#pragma unroll
        for (int k = 0; k < 256; k += 32) {
          v16h a = load_fragA(xrow + k, 256, lane);
          v16h b = load_fragB(wi + k, 256, lane);
          aX = __builtin_amdgcn_wmma_f32_16x16x32_f16(false, a, false, b, (short)0, aX, false, false);
        }
#pragma unroll
        for (int k = 0; k < 256; k += 32) {
          v16h a = load_fragA(hx + k, 256, lane);
          v16h b = load_fragB(wh + k, 256, lane);
          aH = __builtin_amdgcn_wmma_f32_16x16x32_f16(false, a, false, b, (short)0, aH, false, false);
        }
        float bH = bh[nt * 16 + nlane];
        float bX = bi[nt * 16 + nlane];
#pragma unroll
        for (int i = 0; i < 8; ++i) {
          gS[(mb + i) * 1024 + nt * 16 + nlane]       = aH[i] + bH; // hn  (512..767)
          gS[(mb + i) * 1024 + nt * 16 + 256 + nlane] = aX[i] + bX; // xn  (768..1023)
        }
      }
    }
    __syncthreads();

    // gate phase: thread owns column c for all 16 rows
#pragma unroll 4
    for (int m = 0; m < 16; ++m) {
      float r  = sigmoidf_(gS[m * 1024 + c]);
      float z  = sigmoidf_(gS[m * 1024 + 256 + c]);
      float hn = gS[m * 1024 + 512 + c];
      float xn = gS[m * 1024 + 768 + c];
      float nn = tanhf(xn + r * hn);
      float h  = (1.0f - z) * nn + z * (float)hx[m * 256 + c];
      hx[m * 256 + c] = (_Float16)h;
      y[((size_t)tt * N + (n0 + m)) * ycols + col_off + c] = (_Float16)h;
    }
    __syncthreads();
  }
}

// ---- LayerNorm over last dim (one wave per row), f16 in -> f16 out ----
__global__ void ln_kernel(const _Float16* __restrict__ in, const float* __restrict__ g,
                          const float* __restrict__ b, _Float16* __restrict__ out,
                          int R, int C) {
  int wave = threadIdx.x >> 5, lane = threadIdx.x & 31;
  int r = blockIdx.x * 8 + wave;
  if (r >= R) return;
  const _Float16* row = in + (size_t)r * C;
  float v[16];
  int per = C >> 5;                  // 8 or 16
  float s = 0.f;
  for (int i = 0; i < per; ++i) { v[i] = (float)row[lane + i * 32]; s += v[i]; }
  for (int m = 16; m >= 1; m >>= 1) s += __shfl_xor(s, m, 32);
  float mu = s / (float)C;
  float q = 0.f;
  for (int i = 0; i < per; ++i) { float d = v[i] - mu; q += d * d; }
  for (int m = 16; m >= 1; m >>= 1) q += __shfl_xor(q, m, 32);
  float inv = rsqrtf(q / (float)C + 1e-5f);
  for (int i = 0; i < per; ++i) {
    int cc = lane + i * 32;
    out[(size_t)r * C + cc] = (_Float16)((v[i] - mu) * inv * g[cc] + b[cc]);
  }
}

// ---- WMMA GEMM: C(f16, MxNout) = act(A(f16,MxK) @ W(f16,Nout x K)^T + bias)
// block = 8 waves, wave computes 16x64; grid = (Nout/64, M/128)
__global__ void wmma_gemm(const _Float16* __restrict__ A, const _Float16* __restrict__ W,
                          const float* __restrict__ bias, _Float16* __restrict__ C,
                          int M, int K, int Nout, int act) {
  int lane = threadIdx.x & 31, wave = threadIdx.x >> 5;
  size_t m0 = (size_t)blockIdx.y * 128 + (size_t)wave * 16;
  int n0 = blockIdx.x * 64;
  v8f acc[4] = {};
  for (int k = 0; k < K; k += 32) {
    __builtin_prefetch(A + m0 * K + k + 128, 0, 1);   // global_prefetch_b8
    v16h a = load_fragA(A + m0 * K + k, K, lane);
#pragma unroll
    for (int j = 0; j < 4; ++j) {
      v16h b = load_fragB(W + (size_t)(n0 + 16 * j) * K + k, K, lane);
      acc[j] = __builtin_amdgcn_wmma_f32_16x16x32_f16(false, a, false, b, (short)0, acc[j], false, false);
    }
  }
  int n = lane & 15, mb = (lane < 16) ? 0 : 8;
#pragma unroll
  for (int j = 0; j < 4; ++j) {
    float bv = bias[n0 + 16 * j + n];
#pragma unroll
    for (int i = 0; i < 8; ++i) {
      float v = acc[j][i] + bv;
      if (act) v = tanhf(v);
      C[(m0 + mb + i) * Nout + n0 + 16 * j + n] = (_Float16)v;
    }
  }
}

// ---- mean-pool over T: store[(s*32+b)*256+h] = mean_t y3[t, s*B+b, h] ----
__global__ void pool_kernel(const _Float16* __restrict__ y3, _Float16* __restrict__ store) {
  int id = blockIdx.x * 256 + threadIdx.x;   // over S_*B_*256
  int h = id & 255;
  int n = id >> 8;                           // s*B_+b
  float s = 0.f;
  for (int t = 0; t < T_; ++t) s += (float)y3[((size_t)t * N_ + n) * 256 + h];
  store[id] = (_Float16)(s * (1.0f / (float)T_));
}

// ---- final: out[(s*B+b)*2+o] = sigmoid(dot(yln[row,0:512], outW[o,:]) + outb[o]) ----
__global__ void final_kernel(const _Float16* __restrict__ yln, const float* __restrict__ outW,
                             const float* __restrict__ outb, float* __restrict__ out) {
  int id = blockIdx.x * 256 + threadIdx.x;   // over S_*B_*2 = 6144
  if (id >= S_ * B_ * 2) return;
  int o = id & 1;
  int n = id >> 1;
  float acc = outb[o];
  for (int c2 = 0; c2 < 512; ++c2)
    acc += (float)yln[(size_t)n * 512 + c2] * outW[o * 512 + c2];
  out[id] = sigmoidf_(acc);
}

extern "C" void kernel_launch(void* const* d_in, const int* in_sizes, int n_in,
                              void* d_out, int out_size, void* d_ws, size_t ws_size,
                              hipStream_t stream) {
  (void)in_sizes; (void)n_in; (void)out_size; (void)ws_size;
  const int*   tokens  = (const int*)d_in[0];
  const float* emb     = (const float*)d_in[1];
  const float* s_WiF   = (const float*)d_in[2];
  const float* s_WhF   = (const float*)d_in[3];
  const float* s_biF   = (const float*)d_in[4];
  const float* s_bhF   = (const float*)d_in[5];
  const float* s_lnF_g = (const float*)d_in[6];
  const float* s_lnF_b = (const float*)d_in[7];
  const float* s_Wi_f  = (const float*)d_in[8];
  const float* s_Wh_f  = (const float*)d_in[9];
  const float* s_bi_f  = (const float*)d_in[10];
  const float* s_bh_f  = (const float*)d_in[11];
  const float* s_Wi_b  = (const float*)d_in[12];
  const float* s_Wh_b  = (const float*)d_in[13];
  const float* s_bi_b  = (const float*)d_in[14];
  const float* s_bh_b  = (const float*)d_in[15];
  const float* s_ln_g  = (const float*)d_in[16];
  const float* s_ln_b  = (const float*)d_in[17];
  const float* s_linW  = (const float*)d_in[18];
  const float* s_linb  = (const float*)d_in[19];
  const float* dWiF    = (const float*)d_in[20];
  const float* dWhF    = (const float*)d_in[21];
  const float* d_biF   = (const float*)d_in[22];
  const float* d_bhF   = (const float*)d_in[23];
  const float* d_lnF_g = (const float*)d_in[24];
  const float* d_lnF_b = (const float*)d_in[25];
  const float* dWi_f   = (const float*)d_in[26];
  const float* dWh_f   = (const float*)d_in[27];
  const float* d_bi_f  = (const float*)d_in[28];
  const float* d_bh_f  = (const float*)d_in[29];
  const float* dWi_b   = (const float*)d_in[30];
  const float* dWh_b   = (const float*)d_in[31];
  const float* d_bi_b  = (const float*)d_in[32];
  const float* d_bh_b  = (const float*)d_in[33];
  const float* d_ln_g  = (const float*)d_in[34];
  const float* d_ln_b  = (const float*)d_in[35];
  const float* out_W   = (const float*)d_in[36];
  const float* out_b   = (const float*)d_in[37];
  float* out = (float*)d_out;

  // bump allocator over workspace (halfs), 256B-aligned chunks
  _Float16* base = (_Float16*)d_ws;
  size_t off = 0;
  auto alloc = [&](size_t nh) { _Float16* p = base + off; off += (nh + 127) & ~(size_t)127; return p; };

  const size_t WG = 768 * 256;       // GRU weight matrix (3H x 256) in halfs
  _Float16* wSWiF = alloc(WG); _Float16* wSWhF = alloc(WG);
  _Float16* wSWif = alloc(WG); _Float16* wSWhf = alloc(WG);
  _Float16* wSWib = alloc(WG); _Float16* wSWhb = alloc(WG);
  _Float16* wLin  = alloc(256 * 512);
  _Float16* wDWiF = alloc(WG); _Float16* wDWhF = alloc(WG);
  _Float16* wDWif = alloc(WG); _Float16* wDWhf = alloc(WG);
  _Float16* wDWib = alloc(WG); _Float16* wDWhb = alloc(WG);

  const size_t RS = (size_t)T_ * N_;                  // 98304 sentence rows
  _Float16* Xe    = alloc(RS * 256);
  _Float16* y1    = alloc(RS * 256);
  _Float16* y2    = alloc(RS * 512);
  _Float16* y2ln  = alloc(RS * 512);
  _Float16* y1ln  = Xe;    // alias: Xe dead after first scan
  _Float16* y3    = y1;    // alias: y1 dead after LN1
  _Float16* store = alloc((size_t)S_ * B_ * 256);
  _Float16* yd1   = alloc((size_t)S_ * B_ * 256);
  _Float16* yd1ln = alloc((size_t)S_ * B_ * 256);
  _Float16* yd2   = alloc((size_t)S_ * B_ * 512);
  _Float16* yd2ln = alloc((size_t)S_ * B_ * 512);

  auto cast = [&](const float* src, _Float16* dst, int n) {
    cast_kernel<<<(n + 255) / 256, 256, 0, stream>>>(src, dst, n);
  };
  cast(s_WiF, wSWiF, (int)WG);  cast(s_WhF, wSWhF, (int)WG);
  cast(s_Wi_f, wSWif, (int)WG); cast(s_Wh_f, wSWhf, (int)WG);
  cast(s_Wi_b, wSWib, (int)WG); cast(s_Wh_b, wSWhb, (int)WG);
  cast(s_linW, wLin, 256 * 512);
  cast(dWiF, wDWiF, (int)WG);   cast(dWhF, wDWhF, (int)WG);
  cast(dWi_f, wDWif, (int)WG);  cast(dWh_f, wDWhf, (int)WG);
  cast(dWi_b, wDWib, (int)WG);  cast(dWh_b, wDWhb, (int)WG);

  // embedding
  embed_kernel<<<(int)(RS * 256 / 256), 256, 0, stream>>>(tokens, emb, Xe);

  const int scanLds = 16 * 1024 * 4 + 16 * 256 * 2;   // 73728 B dynamic LDS
  dim3 blk(256);

  // --- sentence encoder ---
  gru_scan<<<dim3(N_ / 16), blk, scanLds, stream>>>(Xe, wSWiF, wSWhF, s_biF, s_bhF,
                                                    y1, T_, N_, 256, 0, 0);
  ln_kernel<<<(int)(RS / 8), 256, 0, stream>>>(y1, s_lnF_g, s_lnF_b, y1ln, (int)RS, 256);
  gru_scan<<<dim3(N_ / 16), blk, scanLds, stream>>>(y1ln, wSWif, wSWhf, s_bi_f, s_bh_f,
                                                    y2, T_, N_, 512, 0, 0);
  gru_scan<<<dim3(N_ / 16), blk, scanLds, stream>>>(y1ln, wSWib, wSWhb, s_bi_b, s_bh_b,
                                                    y2, T_, N_, 512, 256, 1);
  ln_kernel<<<(int)(RS / 8), 256, 0, stream>>>(y2, s_ln_g, s_ln_b, y2ln, (int)RS, 512);
  wmma_gemm<<<dim3(256 / 64, (unsigned)(RS / 128)), blk, 0, stream>>>(y2ln, wLin, s_linb,
                                                                     y3, (int)RS, 512, 256, 1);
  pool_kernel<<<(S_ * B_ * 256) / 256, 256, 0, stream>>>(y3, store);

  // --- document encoder (seq=S_, batch=B_) ---
  gru_scan<<<dim3(B_ / 16), blk, scanLds, stream>>>(store, wDWiF, wDWhF, d_biF, d_bhF,
                                                    yd1, S_, B_, 256, 0, 0);
  ln_kernel<<<(S_ * B_) / 8, 256, 0, stream>>>(yd1, d_lnF_g, d_lnF_b, yd1ln, S_ * B_, 256);
  gru_scan<<<dim3(B_ / 16), blk, scanLds, stream>>>(yd1ln, wDWif, wDWhf, d_bi_f, d_bh_f,
                                                    yd2, S_, B_, 512, 0, 0);
  gru_scan<<<dim3(B_ / 16), blk, scanLds, stream>>>(yd1ln, wDWib, wDWhb, d_bi_b, d_bh_b,
                                                    yd2, S_, B_, 512, 256, 1);
  ln_kernel<<<(S_ * B_) / 8, 256, 0, stream>>>(yd2, d_ln_g, d_ln_b, yd2ln, S_ * B_, 512);
  final_kernel<<<(S_ * B_ * 2 + 255) / 256, 256, 0, stream>>>(yd2ln, out_W, out_b, out);
}
